// SelfAttention_18004502905231
// MI455X (gfx1250) — matrix operand order repaired and checked by reading, not compiled, not run on hardware
//
#include <hip/hip_runtime.h>

// ---------------------------------------------------------------------------
// Self-attention (B=4, C=256, H=W=64, CK=32, CV=CO=256) for gfx1250 (CDNA5).
// bf16 WMMA (v_wmma_f32_16x16x32_bf16) for all GEMMs; flash-style 2-pass
// softmax over the query axis; Wo projection fused into the attention kernel.
// ---------------------------------------------------------------------------

#define BATCH 4
#define CCH   256     // input channels C
#define NPIX  4096    // H*W
#define CKEY  32      // key channels

typedef __bf16 v16bf __attribute__((ext_vector_type(16)));
typedef __bf16 v8bf  __attribute__((ext_vector_type(8)));
typedef float  v8f   __attribute__((ext_vector_type(8)));

__device__ __forceinline__ v8f v8f_zero() {
  v8f z;
#pragma unroll
  for (int i = 0; i < 8; ++i) z[i] = 0.0f;
  return z;
}

__device__ __forceinline__ v8f wmma_bf16(v16bf a, v16bf b, v8f c) {
  // D = A(16x32) * B(32x16) + C   (f32 accum)
  return __builtin_amdgcn_wmma_f32_16x16x32_bf16(
      false, a, false, b, (short)0, c, false, false);
}

// A-fragment (16x32, 16-bit): lane L -> row (L&15).
// element i in 0..15 : k = 16*half + i       (16 contiguous k-values)
// `p` points at this lane's 32 contiguous k-values (bf16).
__device__ __forceinline__ v16bf make_afrag(const __bf16* p, int half) {
  v8bf lo = *(const v8bf*)(p + 16 * half);
  v8bf hi = *(const v8bf*)(p + 16 * half + 8);
  v16bf a;
#pragma unroll
  for (int i = 0; i < 8; ++i) { a[i] = lo[i]; a[i + 8] = hi[i]; }
  return a;
}

// B-fragment (32x16, 16-bit): lane L -> column (L&15), k = 16*half + i
// (16 contiguous k-values).  `p` = this column's 32 contiguous k-values.
__device__ __forceinline__ v16bf make_bfrag(const __bf16* p, int half) {
  return *(const v16bf*)(p + 16 * half);
}

// ---------------------------------------------------------------------------
// Projection kernels: 1x1 conv == channel matmul.  x is [B][C][N] f32.
// Q/K stored transposed as [B][N][32] bf16 (k contiguous -> fast fragments).
// V stored as [B][256][N] bf16 (n contiguous -> fast A fragments).
// ---------------------------------------------------------------------------
__global__ void proj_qk_kernel(const float* __restrict__ x,
                               const float* __restrict__ W,
                               const float* __restrict__ bias,
                               __bf16* __restrict__ outT) {
  int idx = blockIdx.x * blockDim.x + threadIdx.x;   // B*N*32 threads
  int r = idx & 31;
  int n = (idx >> 5) & (NPIX - 1);
  int b = idx >> 17;
  float acc = bias[r];
  const float* xr = x + (size_t)b * CCH * NPIX + n;
  const float* wr = W + r * CCH;
#pragma unroll 4
  for (int c = 0; c < CCH; ++c) acc += wr[c] * xr[(size_t)c * NPIX];
  outT[((size_t)b * NPIX + n) * CKEY + r] = (__bf16)acc;
}

__global__ void proj_v_kernel(const float* __restrict__ x,
                              const float* __restrict__ Wv,
                              const float* __restrict__ bv,
                              __bf16* __restrict__ outV) {
  int idx = blockIdx.x * blockDim.x + threadIdx.x;   // B*256*N threads
  int n = idx & (NPIX - 1);
  int v = (idx >> 12) & 255;
  int b = idx >> 20;
  float acc = bv[v];
  const float* xr = x + (size_t)b * CCH * NPIX + n;
  const float* wr = Wv + v * CCH;
#pragma unroll 4
  for (int c = 0; c < CCH; ++c) acc += wr[c] * xr[(size_t)c * NPIX];
  outV[((size_t)b * 256 + v) * NPIX + n] = (__bf16)acc;
}

// One-time f32 -> bf16 copy of Wo (so attn blocks load bf16 A-frags directly).
__global__ void cvt_wo_kernel(const float* __restrict__ Wo,
                              __bf16* __restrict__ Wob) {
  int idx = blockIdx.x * blockDim.x + threadIdx.x;   // 256*256 threads
  Wob[idx] = (__bf16)Wo[idx];
}

// ---------------------------------------------------------------------------
// Pass 1: per-row (n) softmax statistics over m.
// sim[n,m] = sum_k K[n,k]*Q[m,k].  Block = 1 wave, one 16-row n-tile.
// ---------------------------------------------------------------------------
__global__ __launch_bounds__(32)
void rowstats_kernel(const __bf16* __restrict__ Qt,
                     const __bf16* __restrict__ Kt,
                     float* __restrict__ rowmax,
                     float* __restrict__ rowrcp) {
  const int lane = threadIdx.x, half = lane >> 4, l15 = lane & 15;
  const int n0 = blockIdx.x * 16;
  const int b  = blockIdx.y;
  const __bf16* Qb = Qt + (size_t)b * NPIX * CKEY;
  const __bf16* Kb = Kt + (size_t)b * NPIX * CKEY;

  const v16bf ka = make_afrag(Kb + (size_t)(n0 + l15) * CKEY, half);

  float M[8];
#pragma unroll
  for (int r = 0; r < 8; ++r) M[r] = -1e30f;

  // ---- max pass ----
  for (int m0 = 0; m0 < NPIX; m0 += 16) {
    v16bf qb = make_bfrag(Qb + (size_t)(m0 + l15) * CKEY, half);
    v8f d = wmma_bf16(ka, qb, v8f_zero());
#pragma unroll
    for (int r = 0; r < 8; ++r) M[r] = fmaxf(M[r], d[r]);
  }
#pragma unroll
  for (int off = 1; off < 16; off <<= 1)
#pragma unroll
    for (int r = 0; r < 8; ++r) M[r] = fmaxf(M[r], __shfl_xor(M[r], off, 32));

  // ---- sum pass ----
  float S[8];
#pragma unroll
  for (int r = 0; r < 8; ++r) S[r] = 0.0f;
  for (int m0 = 0; m0 < NPIX; m0 += 16) {
    v16bf qb = make_bfrag(Qb + (size_t)(m0 + l15) * CKEY, half);
    v8f d = wmma_bf16(ka, qb, v8f_zero());
#pragma unroll
    for (int r = 0; r < 8; ++r) S[r] += __expf(d[r] - M[r]);
  }
#pragma unroll
  for (int off = 1; off < 16; off <<= 1)
#pragma unroll
    for (int r = 0; r < 8; ++r) S[r] += __shfl_xor(S[r], off, 32);

  if (l15 == 0) {
#pragma unroll
    for (int r = 0; r < 8; ++r) {
      int n = n0 + 8 * half + r;            // D layout: row = r + 8*half
      rowmax[(size_t)b * NPIX + n] = M[r];
      rowrcp[(size_t)b * NPIX + n] = 1.0f / S[r];
    }
  }
}

// ---------------------------------------------------------------------------
// Pass 2: per 16-column m-tile, recompute sim, normalize, accumulate
// ctx[256 x 16] = V * P, then apply Wo, bias, gamma, residual.
// ---------------------------------------------------------------------------
__global__ __launch_bounds__(32)
void attn_kernel(const __bf16* __restrict__ Qt,
                 const __bf16* __restrict__ Kt,
                 const __bf16* __restrict__ Vv,
                 const float* __restrict__ rowmax,
                 const float* __restrict__ rowrcp,
                 const float* __restrict__ x,
                 const __bf16* __restrict__ Wob,
                 const float* __restrict__ bo,
                 const float* __restrict__ gamma,
                 float* __restrict__ out) {
  const int lane = threadIdx.x, half = lane >> 4, l15 = lane & 15;
  const int m0 = blockIdx.x * 16;
  const int b  = blockIdx.y;
  const __bf16* Qb = Qt + (size_t)b * NPIX * CKEY;
  const __bf16* Kb = Kt + (size_t)b * NPIX * CKEY;
  const __bf16* Vb = Vv + (size_t)b * 256 * NPIX;

  __shared__ __align__(64) __bf16 Plds[16][32];     // [m_local][n_local]
  __shared__ __align__(64) __bf16 ctxlds[16][256];  // [m_local][v], bf16
  __shared__ __align__(16) float smax[NPIX];
  __shared__ __align__(16) float srcp[NPIX];

  // Preload this batch's row stats into LDS (vectorized float4).
  {
    const float4* rm = (const float4*)(rowmax + (size_t)b * NPIX);
    const float4* rr = (const float4*)(rowrcp + (size_t)b * NPIX);
    float4* sm = (float4*)smax;
    float4* sr = (float4*)srcp;
    for (int i = lane; i < NPIX / 4; i += 32) {
      sm[i] = rm[i];
      sr[i] = rr[i];
    }
  }
  __syncthreads();

  // Q B-fragment is fixed for the whole block.
  const v16bf qfrag = make_bfrag(Qb + (size_t)(m0 + l15) * CKEY, half);

  v8f ctx[16];
#pragma unroll
  for (int vt = 0; vt < 16; ++vt) ctx[vt] = v8f_zero();

  for (int nc = 0; nc < NPIX; nc += 32) {
    // sim tiles for n-chunk (two 16-row tiles)
    v8f s0 = wmma_bf16(make_afrag(Kb + (size_t)(nc + l15) * CKEY, half),
                       qfrag, v8f_zero());
    v8f s1 = wmma_bf16(make_afrag(Kb + (size_t)(nc + 16 + l15) * CKEY, half),
                       qfrag, v8f_zero());
    // P = exp(sim - rmax[n]) / rsum[n]; stage to LDS in [m][n] order so the
    // next WMMA's B-fragment is a single contiguous 32B ds_load.
#pragma unroll
    for (int r = 0; r < 8; ++r) {
      int n0i = nc + 8 * half + r;
      int n1i = nc + 16 + 8 * half + r;
      Plds[l15][8 * half + r]      = (__bf16)(__expf(s0[r] - smax[n0i]) * srcp[n0i]);
      Plds[l15][16 + 8 * half + r] = (__bf16)(__expf(s1[r] - smax[n1i]) * srcp[n1i]);
    }
    __syncthreads();
    v16bf pfrag = *(const v16bf*)(&Plds[l15][16 * half]);
    // ctx[v, m] += V[v, n-chunk] * P[n-chunk, m]
#pragma unroll
    for (int vt = 0; vt < 16; ++vt) {
      const __bf16* vp = Vb + (size_t)(vt * 16 + l15) * NPIX + nc;
      ctx[vt] = wmma_bf16(make_afrag(vp, half), pfrag, ctx[vt]);
    }
    __syncthreads();
  }

  // Spill ctx to LDS in [m][v] order, already bf16 (WMMA consumes bf16).
#pragma unroll
  for (int vt = 0; vt < 16; ++vt)
#pragma unroll
    for (int r = 0; r < 8; ++r)
      ctxlds[l15][vt * 16 + 8 * half + r] = (__bf16)ctx[vt][r];
  __syncthreads();

  const float g = gamma[0];
  // out[o, m] = gamma * (Wo * ctx + bo) + x
#pragma unroll 1
  for (int ot = 0; ot < 16; ++ot) {
    v8f acc = v8f_zero();
#pragma unroll
    for (int ch = 0; ch < 8; ++ch) {
      // A: Wo rows o = ot*16 + (L&15), v-slice = ch*32.. (bf16, contiguous)
      v16bf aw = make_afrag(Wob + (size_t)(ot * 16 + l15) * 256 + ch * 32, half);
      // B: ctx column m = L&15, 16 contiguous bf16 (32 B) per lane
      v16bf cb = *(const v16bf*)(&ctxlds[l15][ch * 32 + 16 * half]);
      acc = wmma_bf16(aw, cb, acc);
    }
#pragma unroll
    for (int r = 0; r < 8; ++r) {
      int o = ot * 16 + 8 * half + r;
      int m = m0 + l15;
      size_t off = ((size_t)b * 256 + o) * NPIX + m;
      out[off] = g * (acc[r] + bo[o]) + x[off];
    }
  }
}

// ---------------------------------------------------------------------------
extern "C" void kernel_launch(void* const* d_in, const int* in_sizes, int n_in,
                              void* d_out, int out_size, void* d_ws, size_t ws_size,
                              hipStream_t stream) {
  const float* x     = (const float*)d_in[0];
  const float* Wq    = (const float*)d_in[1];
  const float* bq    = (const float*)d_in[2];
  const float* Wk    = (const float*)d_in[3];
  const float* bk    = (const float*)d_in[4];
  const float* Wv    = (const float*)d_in[5];
  const float* bv    = (const float*)d_in[6];
  const float* Wo    = (const float*)d_in[7];
  const float* bo    = (const float*)d_in[8];
  const float* gamma = (const float*)d_in[9];
  float* out = (float*)d_out;

  char* ws = (char*)d_ws;
  __bf16* Qt = (__bf16*)(ws);                          // [B][N][32]   1 MB
  __bf16* Kt = (__bf16*)(ws + (1ull << 20));           // [B][N][32]   1 MB
  __bf16* Vv = (__bf16*)(ws + (2ull << 20));           // [B][256][N]  8 MB
  float* rowmax = (float*)(ws + (10ull << 20));                      // 64 KB
  float* rowrcp = (float*)(ws + (10ull << 20) + (1ull << 16));       // 64 KB
  __bf16* Wob   = (__bf16*)(ws + (10ull << 20) + (2ull << 16));      // 128 KB

  // Projections (Q, K transposed-bf16; V row-major-bf16) + Wo bf16 copy.
  proj_qk_kernel<<<(BATCH * NPIX * CKEY) / 256, 256, 0, stream>>>(x, Wq, bq, Qt);
  proj_qk_kernel<<<(BATCH * NPIX * CKEY) / 256, 256, 0, stream>>>(x, Wk, bk, Kt);
  proj_v_kernel<<<(BATCH * 256 * NPIX) / 256, 256, 0, stream>>>(x, Wv, bv, Vv);
  cvt_wo_kernel<<<(256 * 256) / 256, 256, 0, stream>>>(Wo, Wob);

  dim3 grid(NPIX / 16, BATCH);
  rowstats_kernel<<<grid, 32, 0, stream>>>(Qt, Kt, rowmax, rowrcp);
  attn_kernel<<<grid, 32, 0, stream>>>(Qt, Kt, Vv, rowmax, rowrcp,
                                       x, Wob, bo, gamma, out);
}
